// StandardGRU_34239479283969
// MI455X (gfx1250) — matrix operand order repaired and checked by reading, not compiled, not run on hardware
//
#include <hip/hip_runtime.h>
#include <hip/hip_bf16.h>
#include <math.h>

// ---------------------------------------------------------------------------
// Problem constants
// ---------------------------------------------------------------------------
#define VOCAB 50257
#define DEMB  256
#define HDIM  512
#define BATCH 64
#define SEQ   512
#define H3    1536

#define VPAD  50688            // vocab padded to 99 * 512 (WMMA N-tiling)
#define NT_WD (VPAD / 16)      // 3168 N-tiles for Wd
#define KT_U  (HDIM / 32)      // 16 K-tiles (K=512, bf16 wmma K=32)
#define KT_W  (DEMB / 32)      // 8  K-tiles (K=256)
#define NT_3H (H3 / 16)        // 96 N-tiles over 3H

#define CHUNK_ELEMS (4 * KT_U * 512)   // 32768 bf16 = one gate chunk (4 N-tiles)

typedef __bf16 bf16;
typedef __attribute__((ext_vector_type(16))) __bf16 v16bf;
typedef __attribute__((ext_vector_type(8)))  float  v8f;
typedef unsigned int v4u __attribute__((ext_vector_type(4)));
typedef int          v8i __attribute__((ext_vector_type(8)));
typedef int          v4i __attribute__((ext_vector_type(4)));

__device__ __forceinline__ v8f wmma_bf16(v16bf a, v16bf b, v8f c) {
    return __builtin_amdgcn_wmma_f32_16x16x32_bf16(
        false, a, false, b, (short)0, c, false, false);
}

__device__ __forceinline__ float sigmoidf_(float x) { return 1.0f / (1.0f + expf(-x)); }

// ---------------------------------------------------------------------------
// Init: zero f32 h double-buffer, bf16 packed-A h double-buffer, barrier ctr.
// ---------------------------------------------------------------------------
__global__ void k_init(float* __restrict__ hbuf, unsigned* __restrict__ hpk_u32,
                       unsigned* __restrict__ bar) {
    int i = blockIdx.x * blockDim.x + threadIdx.x;
    if (i < 2 * BATCH * HDIM) hbuf[i] = 0.0f;
    if (i < BATCH * HDIM) hpk_u32[i] = 0u;     // 2 bufs * 64*512 bf16 = 32768 u32 words
    if (i == 0) *bar = 0u;
}

// ---------------------------------------------------------------------------
// Pack a row-major [K, N] fp32 matrix into bf16 WMMA-B operand tiles.
// Tile (nt, kt): 32 lanes x 16 bf16 contiguous per lane.
// B layout (16-bit, 32x16): lane holds col n=lane%16; k = kt*32 + 8*(lane>=16) + e + 8*(e>=8).
// ---------------------------------------------------------------------------
__global__ void k_pack_kxn(const float* __restrict__ src, bf16* __restrict__ dst,
                           int ntiles, int ktiles, int ldn, int nmax) {
    int idx = blockIdx.x * blockDim.x + threadIdx.x;
    int total = ntiles * ktiles * 512;
    if (idx >= total) return;
    int nt  = idx / (ktiles * 512);
    int rem = idx % (ktiles * 512);
    int kt  = rem / 512;
    int r2  = rem % 512;
    int lane = r2 >> 4, e = r2 & 15;
    int n = nt * 16 + (lane & 15);
    int k = kt * 32 + ((lane < 16) ? 0 : 8) + e + ((e >= 8) ? 8 : 0);
    float v = (n < nmax) ? src[(size_t)k * ldn + n] : 0.0f;
    dst[idx] = (bf16)v;
}

// A-operand pack index (16-bit A 16x32, ISA 7.12.2):
// lane<16 : M=lane,    K in {0..7} U {16..23};  lane>=16: M=lane-16, K in {8..15} U {24..31}
__device__ __forceinline__ int a_k_of(int lane, int e) {
    return ((lane < 16) ? 0 : 8) + e + ((e >= 8) ? 8 : 0);
}

// ---------------------------------------------------------------------------
// Phase 1: xp[b,s,:] = emb[tokens[b,s]] @ W + b0   (bf16 WMMA, f32 out)
// ---------------------------------------------------------------------------
__global__ void k_xproj(const int* __restrict__ tokens, const float* __restrict__ emb,
                        const bf16* __restrict__ Wpk, const float* __restrict__ bvec,
                        float* __restrict__ xp) {
    __shared__ int toks[16];
    __shared__ alignas(32) bf16 lA[KT_W * 512];    // 8 KB: 16x256 bf16, A-packed
    const int tid = threadIdx.x;
    const int r0  = blockIdx.x * 16;
    if (tid < 16) toks[tid] = tokens[r0 + tid];
    __syncthreads();
    for (int idx = tid; idx < KT_W * 512; idx += blockDim.x) {
        int kt = idx >> 9, r2 = idx & 511;
        int lane_ = r2 >> 4, e = r2 & 15;
        int m = lane_ & 15;
        int k = kt * 32 + a_k_of(lane_, e);
        lA[idx] = (bf16)emb[(size_t)toks[m] * DEMB + k];
    }
    __syncthreads();
    const int wave = tid >> 5, lane = tid & 31;
    const int mrow = (lane < 16) ? 0 : 8;
    for (int i = 0; i < 6; ++i) {
        int nt = blockIdx.y * 24 + wave * 6 + i;
        v8f acc = {};
        for (int kt = 0; kt < KT_W; ++kt) {
            v16bf a  = *(const v16bf*)(&lA[(kt * 32 + lane) * 16]);
            v16bf bm = *(const v16bf*)(Wpk + ((size_t)(nt * KT_W + kt) * 32 + lane) * 16);
            acc = wmma_bf16(a, bm, acc);
        }
        int n = nt * 16 + (lane & 15);
        float b0 = bvec[n];
        for (int vr = 0; vr < 8; ++vr) {
            int R = r0 + vr + mrow;
            xp[(size_t)R * H3 + n] = acc[vr] + b0;
        }
    }
}

// ---------------------------------------------------------------------------
// Phase 2: persistent GRU scan.
//  - 32 WGs (4 m-tiles x 8 column chunks of 64), 4 waves each.
//  - Per-WG U slice (192 KB) staged ONCE into dynamic LDS via TDM
//    (tensor_load_to_lds, 2-D descriptor: 3 rows x 64 KB, 512 KB stride).
//  - h kept twice: f32 row-major (gate math) + bf16 A-packed (WMMA operand),
//    both double-buffered in global/L2; packed writes are direct scatter.
//  - software grid barrier per step.
// ---------------------------------------------------------------------------
__global__ void k_scan(const int* __restrict__ tokens, const float* __restrict__ xp,
                       const bf16* __restrict__ Upk, const float* __restrict__ bvec,
                       float* __restrict__ hbuf, bf16* __restrict__ hpk,
                       unsigned* __restrict__ bar) {
    extern __shared__ bf16 lU[];                   // 3 * 32768 bf16 = 192 KB
    const int tid  = threadIdx.x;
    const int wg   = blockIdx.x;                   // 0..31
    const int mt   = wg >> 3;
    const int m0   = mt * 16;
    const int jc   = wg & 7;
    const int wave = tid >> 5, lane = tid & 31;
    const int jsub = jc * 64 + wave * 16;
    const int mrow = (lane < 16) ? 0 : 8;
    const int nH   = jsub + (lane & 15);
    const float b1z = bvec[H3 + nH];
    const float b1r = bvec[H3 + 512 + nH];
    const float b1h = bvec[H3 + 1024 + nH];
    // packed-A scatter coords for this lane's column nH (constants across steps)
    const int kt0 = nH >> 5;
    const int r32 = nH & 31;
    const int hiA = (r32 >> 3) & 1;
    const int eA  = r32 - 8 * hiA - ((r32 >= 16) ? 8 : 0);

    // ---- stage U slice into LDS (once) ----
    const size_t c0 = ((size_t)(jc * 4) * KT_U) * 512;   // gate-z chunk base (elems)
#if defined(__has_builtin) && __has_builtin(__builtin_amdgcn_tensor_load_to_lds) && __has_builtin(__builtin_amdgcn_s_wait_tensorcnt)
    if (tid < 32) {
        unsigned long long ga = (unsigned long long)(const void*)(Upk + c0);
        unsigned lds0 = (unsigned)(unsigned long long)(void*)lU;
        v4u g0;
        g0[0] = 1u;                                    // count=1, user descriptor
        g0[1] = lds0;                                  // lds_addr
        g0[2] = (unsigned)(ga & 0xFFFFFFFFu);          // global_addr[31:0]
        g0[3] = (unsigned)((ga >> 32) & 0x1FFFFFFu) | (2u << 30);  // addr[56:32] | type=2
        v8i g1;
        g1[0] = (int)(3u << 16);                       // wg_mask=0, data_size=8B
        g1[1] = (int)(8192u << 16);                    // tensor_dim0=8192 (lo16 @[63:48])
        g1[2] = (int)(3u << 16);                       // tensor_dim0 hi | tensor_dim1=3
        g1[3] = (int)(8192u << 16);                    // tensor_dim1 hi | tile_dim0=8192
        g1[4] = (int)(3u);                             // tile_dim1=3, tile_dim2=0
        g1[5] = (int)65536;                            // tensor_dim0_stride = 512KB/8
        g1[6] = 0;
        g1[7] = 0;
        v4i g2 = {0, 0, 0, 0};
        v4i g3 = {0, 0, 0, 0};
        v8i g4 = {0, 0, 0, 0, 0, 0, 0, 0};             // unused trailing group (6-arg form)
        __builtin_amdgcn_tensor_load_to_lds(g0, g1, g2, g3, g4, 0);
        __builtin_amdgcn_s_wait_tensorcnt(0);
    }
#else
    for (int g = 0; g < 3; ++g) {
        const uint4* s = (const uint4*)(Upk + c0 + (size_t)g * 32 * KT_U * 512);
        uint4*       d = (uint4*)(lU + g * CHUNK_ELEMS);
        for (int i = tid; i < CHUNK_ELEMS / 8; i += 128) d[i] = s[i];
    }
#endif
    __syncthreads();

    const bf16* lUw = lU + wave * (KT_U * 512);    // this wave's N-tile within each chunk

    for (int t = 0; t < SEQ; ++t) {
        const int p = t & 1;
        const float* hr   = hbuf + (size_t)p * BATCH * HDIM;
        float*       hw   = hbuf + (size_t)(1 - p) * BATCH * HDIM;
        const bf16*  Apk  = hpk + ((size_t)(p * 4 + mt) * KT_U) * 512;
        bf16*        Apkw = hpk + ((size_t)((1 - p) * 4 + mt) * KT_U) * 512;

        v8f az = {}, ar = {}, ah = {};
        for (int kt = 0; kt < KT_U; ++kt) {
            v16bf a  = *(const v16bf*)(Apk + (size_t)kt * 512 + lane * 16);
            v16bf bz = *(const v16bf*)(lUw + 0 * CHUNK_ELEMS + kt * 512 + lane * 16);
            v16bf br = *(const v16bf*)(lUw + 1 * CHUNK_ELEMS + kt * 512 + lane * 16);
            v16bf bh = *(const v16bf*)(lUw + 2 * CHUNK_ELEMS + kt * 512 + lane * 16);
            az = wmma_bf16(a, bz, az);
            ar = wmma_bf16(a, br, ar);
            ah = wmma_bf16(a, bh, ah);
        }

        // gate combine in registers (D layout: lane col n=lane%16, rows vr+mrow)
        for (int vr = 0; vr < 8; ++vr) {
            int gb = m0 + vr + mrow;
            size_t xb = ((size_t)gb * SEQ + t) * H3;
            float xz = xp[xb + nH];
            float xr = xp[xb + 512 + nH];
            float xh = xp[xb + 1024 + nH];
            float z  = sigmoidf_(xz + az[vr] + b1z);
            float r  = sigmoidf_(xr + ar[vr] + b1r);
            float hh = tanhf(xh + r * (ah[vr] + b1h));   // reset_after=True
            float hold = hr[(size_t)gb * HDIM + nH];
            float hn = z * hold + (1.0f - z) * hh;
            if (tokens[gb * SEQ + t] == 0) hn = hold;    // mask_zero carry-through
            hw[(size_t)gb * HDIM + nH] = hn;             // f32 state
            int laneA = (vr + mrow) + 16 * hiA;          // bf16 A-packed state
            Apkw[(size_t)kt0 * 512 + laneA * 16 + eA] = (bf16)hn;
        }

        // grid barrier (32 co-resident WGs; counter monotonic over steps)
        __threadfence();
        __syncthreads();
        if (tid == 0) {
            atomicAdd(bar, 1u);
            unsigned tgt = 32u * (unsigned)(t + 1);
            while (*((volatile unsigned*)bar) < tgt) { __builtin_amdgcn_s_sleep(2); }
        }
        __syncthreads();
        __threadfence();
    }
}

// ---------------------------------------------------------------------------
// Phase 3: logits = h_last @ Wd + bd  (bf16 WMMA; Wd zero-padded to VPAD cols)
// ---------------------------------------------------------------------------
__global__ void k_logits(const float* __restrict__ hlast, const bf16* __restrict__ Wdpk,
                         const float* __restrict__ bd, float* __restrict__ logits) {
    __shared__ alignas(32) bf16 lA[KT_U * 512];
    const int tid = threadIdx.x;
    const int m0  = blockIdx.x * 16;
    for (int idx = tid; idx < KT_U * 512; idx += blockDim.x) {
        int kt = idx >> 9, r2 = idx & 511;
        int lane_ = r2 >> 4, e = r2 & 15;
        int m = lane_ & 15;
        int k = kt * 32 + a_k_of(lane_, e);
        lA[idx] = (bf16)hlast[(size_t)(m0 + m) * HDIM + k];
    }
    __syncthreads();
    const int wave = tid >> 5, lane = tid & 31;
    const int mrow = (lane < 16) ? 0 : 8;
    for (int i = 0; i < 8; ++i) {
        int nt = blockIdx.y * 32 + wave * 8 + i;
        v8f acc = {};
        for (int kt = 0; kt < KT_U; ++kt) {
            v16bf a  = *(const v16bf*)(&lA[(kt * 32 + lane) * 16]);
            v16bf bm = *(const v16bf*)(Wdpk + ((size_t)(nt * KT_U + kt) * 32 + lane) * 16);
            acc = wmma_bf16(a, bm, acc);
        }
        int n = nt * 16 + (lane & 15);
        float bb = (n < VOCAB) ? bd[n] : 0.0f;
        for (int vr = 0; vr < 8; ++vr)
            logits[(size_t)(m0 + vr + mrow) * VPAD + n] = acc[vr] + bb;
    }
}

// ---------------------------------------------------------------------------
// Phase 4: softmax (online max/sum per row, then normalize)
// ---------------------------------------------------------------------------
__global__ void k_rowstat(const float* __restrict__ lg, float* __restrict__ rm,
                          float* __restrict__ rs) {
    __shared__ float sm[256], ss[256];
    int b = blockIdx.x, tid = threadIdx.x;
    const float* row = lg + (size_t)b * VPAD;
    float m = -INFINITY, s = 0.0f;
    for (int n = tid; n < VOCAB; n += 256) {
        float x = row[n];
        if (x > m) { s = s * expf(m - x) + 1.0f; m = x; }
        else       { s += expf(x - m); }
    }
    sm[tid] = m; ss[tid] = s;
    __syncthreads();
    for (int o = 128; o > 0; o >>= 1) {
        if (tid < o) {
            float m2 = sm[tid + o], s2 = ss[tid + o];
            float M = fmaxf(sm[tid], m2);
            ss[tid] = ss[tid] * expf(sm[tid] - M) + s2 * expf(m2 - M);
            sm[tid] = M;
        }
        __syncthreads();
    }
    if (tid == 0) { rm[b] = sm[0]; rs[b] = ss[0]; }
}

__global__ void k_norm(const float* __restrict__ lg, const float* __restrict__ rm,
                       const float* __restrict__ rs, float* __restrict__ out) {
    int i = blockIdx.x * blockDim.x + threadIdx.x;
    if (i >= BATCH * VOCAB) return;
    int b = i / VOCAB, n = i % VOCAB;
    out[i] = expf(lg[(size_t)b * VPAD + n] - rm[b]) / rs[b];
}

// ---------------------------------------------------------------------------
// Host launcher
// ---------------------------------------------------------------------------
extern "C" void kernel_launch(void* const* d_in, const int* in_sizes, int n_in,
                              void* d_out, int out_size, void* d_ws, size_t ws_size,
                              hipStream_t stream) {
    const int*   tokens = (const int*)  d_in[0];   // [B,S]
    const float* emb    = (const float*)d_in[1];   // [V,D]
    const float* W      = (const float*)d_in[2];   // [D,3H]
    const float* U      = (const float*)d_in[3];   // [H,3H]
    const float* bvec   = (const float*)d_in[4];   // [2,3H]
    const float* Wd     = (const float*)d_in[5];   // [H,V]
    const float* bd     = (const float*)d_in[6];   // [V]
    float* out = (float*)d_out;                    // [B,V]

    char* ws = (char*)d_ws;
    size_t off = 0;
    auto give = [&](size_t bytes) -> size_t {
        size_t o = off;
        off += (bytes + 255) & ~(size_t)255;
        return o;
    };
    size_t o_bar  = give(4);
    size_t o_h    = give((size_t)2 * BATCH * HDIM * 4);          // f32 h double buffer
    size_t o_hpk  = give((size_t)2 * BATCH * HDIM * 2);          // bf16 A-packed h dbuf
    size_t o_rm   = give((size_t)BATCH * 4);
    size_t o_rs   = give((size_t)BATCH * 4);
    size_t o_upk  = give((size_t)NT_3H * KT_U * 512 * 2);        // 1.5 MB bf16
    size_t o_wpk  = give((size_t)NT_3H * KT_W * 512 * 2);        // 0.75 MB
    size_t o_wdpk = give((size_t)NT_WD * KT_U * 512 * 2);        // ~52 MB
    size_t o_lg   = give((size_t)BATCH * VPAD * 4);              // ~13 MB
    size_t o_xp   = give((size_t)BATCH * SEQ * H3 * 4);          // ~192 MB
    (void)ws_size; (void)in_sizes; (void)n_in; (void)out_size;

    unsigned* bar  = (unsigned*)(ws + o_bar);
    float*    hbuf = (float*)(ws + o_h);
    bf16*     hpk  = (bf16*)(ws + o_hpk);
    float*    rowm = (float*)(ws + o_rm);
    float*    rows = (float*)(ws + o_rs);
    bf16*     Upk  = (bf16*)(ws + o_upk);
    bf16*     Wpk  = (bf16*)(ws + o_wpk);
    bf16*     Wdpk = (bf16*)(ws + o_wdpk);
    float*    lgts = (float*)(ws + o_lg);
    float*    xp   = (float*)(ws + o_xp);

    k_init<<<(2 * BATCH * HDIM + 255) / 256, 256, 0, stream>>>(hbuf, (unsigned*)hpk, bar);
    k_pack_kxn<<<(NT_3H * KT_U * 512 + 255) / 256, 256, 0, stream>>>(U,  Upk,  NT_3H, KT_U, H3,    H3);
    k_pack_kxn<<<(NT_3H * KT_W * 512 + 255) / 256, 256, 0, stream>>>(W,  Wpk,  NT_3H, KT_W, H3,    H3);
    k_pack_kxn<<<(NT_WD * KT_U * 512 + 255) / 256, 256, 0, stream>>>(Wd, Wdpk, NT_WD, KT_U, VOCAB, VOCAB);

    k_xproj<<<dim3(BATCH * SEQ / 16, H3 / 384), 128, 0, stream>>>(tokens, emb, Wpk, bvec, xp);
    // 192 KB dynamic LDS: whole per-WG U slice resident for all 512 steps
    k_scan<<<32, 128, 3 * CHUNK_ELEMS * sizeof(bf16), stream>>>(tokens, xp, Upk, bvec, hbuf, hpk, bar);
    // S even => h_last lives in f32 buffer 0
    k_logits<<<dim3(BATCH / 16, VPAD / 512), 128, 0, stream>>>(hbuf, Wdpk, bd, lgts);
    k_rowstat<<<BATCH, 256, 0, stream>>>(lgts, rowm, rows);
    k_norm<<<(BATCH * VOCAB + 255) / 256, 256, 0, stream>>>(lgts, rowm, rows, out);
}
